// MetaGNN_56908316672645
// MI455X (gfx1250) — compile-verified, matrix operand order checked by reading, not run on hardware
//
#include <hip/hip_runtime.h>
#include <hip/hip_bf16.h>

#define N_NODES 50000
#define N_EDGES 600000
#define D_NODE  128
#define D_EDGE  64
#define HDIM    128

typedef __bf16 bf16_t;
typedef bf16_t v16bf __attribute__((ext_vector_type(16)));
typedef float  v8f   __attribute__((ext_vector_type(8)));

union FragAB { v16bf v; unsigned u[8]; };

// K-position of fragment dword i for 16-bit 16x32 A/B operands (ISA 7.12.2)
__device__ __forceinline__ int kmap(int i, int hf) {
  return ((i < 4) ? (2 * i) : (16 + 2 * (i - 4))) + 8 * hf;
}

__device__ __forceinline__ unsigned pack2(float a, float b) {
  union { bf16_t h[2]; unsigned u; } t;
  t.h[0] = (bf16_t)a; t.h[1] = (bf16_t)b;
  return t.u;
}

__device__ __forceinline__ v8f wmma_bf16(const FragAB& a, const FragAB& b, v8f c) {
  return __builtin_amdgcn_wmma_f32_16x16x32_bf16(false, a.v, false, b.v,
                                                 (short)0, c, false, false);
}

__device__ __forceinline__ void wave_lds_fence() {
  __builtin_amdgcn_wave_barrier();
  asm volatile("s_wait_dscnt 0" ::: "memory");
}

// Async global->LDS copy of 16 bytes per lane (CDNA5 GLOBAL_LOAD_ASYNC_TO_LDS_B128,
// tracked by ASYNCcnt). lds_addr: wave-relative LDS byte address (low 32 bits of
// the generic pointer, per ISA 10.2 aperture mapping).
__device__ __forceinline__ void async_copy_b128(unsigned lds_addr, unsigned voff,
                                                unsigned long long gbase) {
  asm volatile("global_load_async_to_lds_b128 %0, %1, %2"
               :: "v"(lds_addr), "v"(voff), "s"(gbase) : "memory");
}
__device__ __forceinline__ void wait_async0() {
  asm volatile("s_wait_asynccnt 0" ::: "memory");
}

// ---------------------------------------------------------------------------
// Edge kernel: e = ea @ W + b ; msg = relu(x[src] + e) ; atomic scatter to aggr
// W: [64,128] row-major. One wave handles a 16-edge x 128-out tile.
// ---------------------------------------------------------------------------
__global__ __launch_bounds__(256) void edge_kernel(
    const float* __restrict__ x,     // [N,128]
    const int*   __restrict__ ei,    // [2,E]
    const float* __restrict__ ea,    // [E,64]
    const float* __restrict__ W,     // [64,128]
    const float* __restrict__ bias,  // [128]
    float*       __restrict__ aggr)  // [N,128]
{
  __shared__ float  Wstage[64 * 128];  // raw f32 weights, async-staged (32 KB)
  __shared__ bf16_t Wt[128 * 64];      // Wt[n*64 + k] bf16 transposed (16 KB)
  __shared__ float  bsh[128];
  const int tid = threadIdx.x;

  // Async-stage the f32 weight matrix into LDS (256 thr * 16 B * 8 iters = 32 KB)
  {
    const unsigned long long gbase = (unsigned long long)(uintptr_t)W;
    const unsigned ldsbase = (unsigned)(uintptr_t)&Wstage[0];
#pragma unroll
    for (int j = 0; j < 8; j++) {
      const unsigned off = (unsigned)(tid + j * 256) * 16u;
      async_copy_b128(ldsbase + off, off, gbase);
    }
    wait_async0();
  }
  __syncthreads();
  // Convert + transpose to bf16 out of LDS
  for (int i = tid; i < 64 * 128; i += 256) {
    int k = i >> 7, n = i & 127;
    Wt[n * 64 + k] = (bf16_t)Wstage[i];
  }
  if (tid < 128) bsh[tid] = bias[tid];
  __syncthreads();

  const int wave = tid >> 5, lane = tid & 31;
  const int hf = lane >> 4, ln = lane & 15;
  const int tiles = N_EDGES / 16;

  for (int tile = blockIdx.x * 8 + wave; tile < tiles; tile += gridDim.x * 8) {
    const int ebase = tile * 16;
    // A fragments: row = edge (ebase+ln), K = 64 in two steps of 32
    const float* ap = ea + (size_t)(ebase + ln) * 64;
    FragAB a0, a1;
#pragma unroll
    for (int i = 0; i < 8; i++) {
      int k = kmap(i, hf);
      a0.u[i] = pack2(ap[k], ap[k + 1]);
      a1.u[i] = pack2(ap[32 + k], ap[32 + k + 1]);
    }
    // Hoist the 8 src/dst indices this lane needs (rows r+8*hf), shifted to
    // row offsets, loop-invariant across the 8 output n-tiles.
    const int er = ebase + 8 * hf;
    const int4 sA = *(const int4*)(ei + er);
    const int4 sB = *(const int4*)(ei + er + 4);
    const int4 dA = *(const int4*)(ei + N_EDGES + er);
    const int4 dB = *(const int4*)(ei + N_EDGES + er + 4);
    int soff[8] = { sA.x << 7, sA.y << 7, sA.z << 7, sA.w << 7,
                    sB.x << 7, sB.y << 7, sB.z << 7, sB.w << 7 };
    int doff[8] = { dA.x << 7, dA.y << 7, dA.z << 7, dA.w << 7,
                    dB.x << 7, dB.y << 7, dB.z << 7, dB.w << 7 };

#pragma unroll
    for (int nt = 0; nt < 8; nt++) {
      const int n = nt * 16 + ln;
      // Issue the 8 gather loads up front so they overlap the LDS B-fragment
      // reads and the two WMMAs below (hides L2 latency).
      float xv[8];
#pragma unroll
      for (int r = 0; r < 8; r++) xv[r] = x[soff[r] + n];

      FragAB b0, b1;
      const bf16_t* wp = &Wt[n * 64];
#pragma unroll
      for (int i = 0; i < 8; i++) {
        int k = kmap(i, hf);
        b0.u[i] = *(const unsigned*)(wp + k);
        b1.u[i] = *(const unsigned*)(wp + 32 + k);
      }
      v8f c;
      const float bv = bsh[n];
#pragma unroll
      for (int r = 0; r < 8; r++) c[r] = bv;
      c = wmma_bf16(a0, b0, c);
      c = wmma_bf16(a1, b1, c);
      // fused relu + scatter-add (skip zero contributions)
#pragma unroll
      for (int r = 0; r < 8; r++) {
        float v = c[r] + xv[r];
        if (v > 0.0f) atomicAdd(&aggr[doff[r] + n], v);
      }
    }
  }
}

// ---------------------------------------------------------------------------
// Fused GINE node MLP: out = relu((x+aggr)@W1 + b1) @ W2 + b2
// W1: [128,256], W2: [256,128]. Back-to-back WMMA chained through LDS staging.
// ---------------------------------------------------------------------------
__global__ __launch_bounds__(256) void node_kernel(
    const float* __restrict__ x,
    const float* __restrict__ aggr,
    const float* __restrict__ W1,   // [128,256]
    const float* __restrict__ b1,   // [256]
    const float* __restrict__ W2,   // [256,128]
    const float* __restrict__ b2,   // [128]
    float*       __restrict__ out)  // [N,128]
{
  __shared__ bf16_t W1t[256 * 128];        // [n][k]  64 KB
  __shared__ bf16_t W2t[128 * 256];        // [n][k]  64 KB
  __shared__ float  b1s[256], b2s[128];
  __shared__ bf16_t stage[8][16 * 32];     // per-wave relu(hidden) tile, 8 KB

  const int tid = threadIdx.x;
  for (int i = tid; i < 128 * 256; i += 256) {
    int k = i >> 8, n = i & 255;
    W1t[n * 128 + k] = (bf16_t)W1[i];
  }
  for (int i = tid; i < 256 * 128; i += 256) {
    int k = i >> 7, n = i & 127;
    W2t[n * 256 + k] = (bf16_t)W2[i];
  }
  b1s[tid] = b1[tid];
  if (tid < 128) b2s[tid] = b2[tid];
  __syncthreads();

  const int wave = tid >> 5, lane = tid & 31;
  const int hf = lane >> 4, ln = lane & 15;
  const int tiles = N_NODES / 16;

  for (int tile = blockIdx.x * 8 + wave; tile < tiles; tile += gridDim.x * 8) {
    const int nbase = tile * 16;
    const float* xp = x + (size_t)(nbase + ln) * 128;
    const float* gp = aggr + (size_t)(nbase + ln) * 128;

    FragAB ain[4];
#pragma unroll
    for (int kk = 0; kk < 4; kk++)
#pragma unroll
      for (int i = 0; i < 8; i++) {
        int k = kk * 32 + kmap(i, hf);
        ain[kk].u[i] = pack2(xp[k] + gp[k], xp[k + 1] + gp[k + 1]);
      }

    v8f acc[8];
#pragma unroll
    for (int nt = 0; nt < 8; nt++) {
      const float bv = b2s[nt * 16 + ln];
#pragma unroll
      for (int r = 0; r < 8; r++) acc[nt][r] = bv;
    }

    for (int hc = 0; hc < 8; hc++) {          // 8 chunks of 32 hidden cols
#pragma unroll
      for (int sub = 0; sub < 2; sub++) {     // two 16-col WMMA sub-chunks
        const int nh = hc * 32 + sub * 16 + ln;
        v8f d;
        const float bv = b1s[nh];
#pragma unroll
        for (int r = 0; r < 8; r++) d[r] = bv;
        const bf16_t* wp = &W1t[nh * 128];
#pragma unroll
        for (int kk = 0; kk < 4; kk++) {
          FragAB bf;
#pragma unroll
          for (int i = 0; i < 8; i++)
            bf.u[i] = *(const unsigned*)(wp + kk * 32 + kmap(i, hf));
          d = wmma_bf16(ain[kk], bf, d);
        }
        // relu + stage to LDS as A-operand of second GEMM
#pragma unroll
        for (int r = 0; r < 8; r++) {
          float v = d[r] > 0.0f ? d[r] : 0.0f;
          stage[wave][(r + 8 * hf) * 32 + sub * 16 + ln] = (bf16_t)v;
        }
      }
      wave_lds_fence();
      FragAB a2;
      const bf16_t* sp = &stage[wave][ln * 32];
#pragma unroll
      for (int i = 0; i < 8; i++)
        a2.u[i] = *(const unsigned*)(sp + kmap(i, hf));
#pragma unroll
      for (int nt = 0; nt < 8; nt++) {
        const int n = nt * 16 + ln;
        FragAB b2f;
        const bf16_t* wp2 = &W2t[n * 256 + hc * 32];
#pragma unroll
        for (int i = 0; i < 8; i++)
          b2f.u[i] = *(const unsigned*)(wp2 + kmap(i, hf));
        acc[nt] = wmma_bf16(a2, b2f, acc[nt]);
      }
      wave_lds_fence();
    }
#pragma unroll
    for (int nt = 0; nt < 8; nt++)
#pragma unroll
      for (int r = 0; r < 8; r++)
        out[(size_t)(nbase + r + 8 * hf) * 128 + nt * 16 + ln] = acc[nt][r];
  }
}

// ---------------------------------------------------------------------------
// BatchNorm: stats (sum, sumsq) -> finalize (scale, shift) -> apply (+relu)
// stats layout: [0:128) sum, [128:256) sumsq, [256:384) scale, [384:512) shift
// ---------------------------------------------------------------------------
__global__ __launch_bounds__(128) void bn_stats(const float* __restrict__ h,
                                                float* __restrict__ stats) {
  const int f = threadIdx.x;
  float s = 0.0f, q = 0.0f;
  for (int r = blockIdx.x; r < N_NODES; r += gridDim.x) {
    float v = h[(size_t)r * 128 + f];
    s += v; q += v * v;
  }
  atomicAdd(&stats[f], s);
  atomicAdd(&stats[128 + f], q);
}

__global__ __launch_bounds__(128) void bn_finalize(float* __restrict__ stats,
                                                   const float* __restrict__ g,
                                                   const float* __restrict__ b) {
  const int f = threadIdx.x;
  const float invn = 1.0f / (float)N_NODES;
  float mu = stats[f] * invn;
  float var = stats[128 + f] * invn - mu * mu;
  float sc = rsqrtf(var + 1e-5f) * g[f];
  stats[256 + f] = sc;
  stats[384 + f] = b[f] - mu * sc;
}

__global__ __launch_bounds__(256) void bn_apply(float* __restrict__ h,
                                                const float* __restrict__ stats) {
  const int total = N_NODES * 128;
  for (int i = blockIdx.x * 256 + threadIdx.x; i < total; i += gridDim.x * 256) {
    const int f = i & 127;
    float v = h[i] * stats[256 + f] + stats[384 + f];
    h[i] = v > 0.0f ? v : 0.0f;
  }
}

// ---------------------------------------------------------------------------
// Readout: out = relu(h @ W1 + b1) @ w2 + b2, W1: [128,512], w2: [512]
// Hidden never materialized: chunk-wise WMMA + in-register dot + shfl reduce.
// ---------------------------------------------------------------------------
__global__ __launch_bounds__(256) void readout_kernel(
    const float* __restrict__ h,    // [N,128]
    const float* __restrict__ W1,   // [128,512]
    const float* __restrict__ b1,   // [512]
    const float* __restrict__ w2,   // [512]
    const float* __restrict__ b2,   // [1]
    float*       __restrict__ out)  // [N]
{
  __shared__ bf16_t W1t[512 * 128];  // [n][k] 128 KB
  __shared__ float  b1s[512], w2s[512];
  __shared__ float  b2sh;
  const int tid = threadIdx.x;
  for (int i = tid; i < 128 * 512; i += 256) {
    int k = i >> 9, n = i & 511;
    W1t[n * 128 + k] = (bf16_t)W1[i];
  }
  for (int i = tid; i < 512; i += 256) { b1s[i] = b1[i]; w2s[i] = w2[i]; }
  if (tid == 0) b2sh = b2[0];
  __syncthreads();

  const int wave = tid >> 5, lane = tid & 31;
  const int hf = lane >> 4, ln = lane & 15;
  const int tiles = N_NODES / 16;

  for (int tile = blockIdx.x * 8 + wave; tile < tiles; tile += gridDim.x * 8) {
    const int nbase = tile * 16;
    const float* hp = h + (size_t)(nbase + ln) * 128;
    FragAB ain[4];
#pragma unroll
    for (int kk = 0; kk < 4; kk++)
#pragma unroll
      for (int i = 0; i < 8; i++) {
        int k = kk * 32 + kmap(i, hf);
        ain[kk].u[i] = pack2(hp[k], hp[k + 1]);
      }
    float oacc[8];
#pragma unroll
    for (int r = 0; r < 8; r++) oacc[r] = 0.0f;

    for (int hc = 0; hc < 32; hc++) {   // 512 hidden cols in 16-wide tiles
      const int n = hc * 16 + ln;
      v8f d;
      const float bv = b1s[n];
#pragma unroll
      for (int r = 0; r < 8; r++) d[r] = bv;
      const bf16_t* wp = &W1t[n * 128];
#pragma unroll
      for (int kk = 0; kk < 4; kk++) {
        FragAB bf;
#pragma unroll
        for (int i = 0; i < 8; i++)
          bf.u[i] = *(const unsigned*)(wp + kk * 32 + kmap(i, hf));
        d = wmma_bf16(ain[kk], bf, d);
      }
      const float wv = w2s[n];
#pragma unroll
      for (int r = 0; r < 8; r++) {
        float v = d[r] > 0.0f ? d[r] : 0.0f;
        oacc[r] += v * wv;
      }
    }
    // reduce the 16 column-lanes of each half-wave
#pragma unroll
    for (int off = 1; off < 16; off <<= 1)
#pragma unroll
      for (int r = 0; r < 8; r++)
        oacc[r] += __shfl_xor(oacc[r], off, 32);
    if (ln == 0) {
#pragma unroll
      for (int r = 0; r < 8; r++)
        out[nbase + r + 8 * hf] = oacc[r] + b2sh;
    }
  }
}

// ---------------------------------------------------------------------------
extern "C" void kernel_launch(void* const* d_in, const int* in_sizes, int n_in,
                              void* d_out, int out_size, void* d_ws, size_t ws_size,
                              hipStream_t stream) {
  const float* x      = (const float*)d_in[0];
  const int*   ei     = (const int*)  d_in[1];
  const float* ea     = (const float*)d_in[2];
  const float* ee_w0  = (const float*)d_in[3];
  const float* ee_b0  = (const float*)d_in[4];
  const float* ee_w1  = (const float*)d_in[5];
  const float* ee_b1  = (const float*)d_in[6];
  const float* nn0_w1 = (const float*)d_in[7];
  const float* nn0_b1 = (const float*)d_in[8];
  const float* nn0_w2 = (const float*)d_in[9];
  const float* nn0_b2 = (const float*)d_in[10];
  const float* nn1_w1 = (const float*)d_in[11];
  const float* nn1_b1 = (const float*)d_in[12];
  const float* nn1_w2 = (const float*)d_in[13];
  const float* nn1_b2 = (const float*)d_in[14];
  const float* bn0_g  = (const float*)d_in[15];
  const float* bn0_b  = (const float*)d_in[16];
  const float* bn1_g  = (const float*)d_in[17];
  const float* bn1_b  = (const float*)d_in[18];
  const float* mlp_w1 = (const float*)d_in[19];
  const float* mlp_b1 = (const float*)d_in[20];
  const float* mlp_w2 = (const float*)d_in[21];
  const float* mlp_b2 = (const float*)d_in[22];
  float* out = (float*)d_out;

  const size_t NF = (size_t)N_NODES * 128;
  float* aggr  = (float*)d_ws;
  float* h0    = aggr + NF;
  float* h1    = h0 + NF;
  float* stats = h1 + NF;   // 512 floats

  const int edgeBlocks = 512;
  const int nodeBlocks = 391;   // 391*8 waves >= 3125 tiles

  // ---- layer 0 ----
  hipMemsetAsync(aggr, 0, NF * sizeof(float), stream);
  edge_kernel<<<edgeBlocks, 256, 0, stream>>>(x, ei, ea, ee_w0, ee_b0, aggr);
  node_kernel<<<nodeBlocks, 256, 0, stream>>>(x, aggr, nn0_w1, nn0_b1, nn0_w2, nn0_b2, h0);
  hipMemsetAsync(stats, 0, 256 * sizeof(float), stream);
  bn_stats<<<512, 128, 0, stream>>>(h0, stats);
  bn_finalize<<<1, 128, 0, stream>>>(stats, bn0_g, bn0_b);
  bn_apply<<<1024, 256, 0, stream>>>(h0, stats);

  // ---- layer 1 ----
  hipMemsetAsync(aggr, 0, NF * sizeof(float), stream);
  edge_kernel<<<edgeBlocks, 256, 0, stream>>>(h0, ei, ea, ee_w1, ee_b1, aggr);
  node_kernel<<<nodeBlocks, 256, 0, stream>>>(h0, aggr, nn1_w1, nn1_b1, nn1_w2, nn1_b2, h1);
  hipMemsetAsync(stats, 0, 256 * sizeof(float), stream);
  bn_stats<<<512, 128, 0, stream>>>(h1, stats);
  bn_finalize<<<1, 128, 0, stream>>>(stats, bn1_g, bn1_b);
  bn_apply<<<1024, 256, 0, stream>>>(h1, stats);

  // ---- readout ----
  readout_kernel<<<nodeBlocks, 256, 0, stream>>>(h1, mlp_w1, mlp_b1, mlp_w2, mlp_b2, out);

  (void)in_sizes; (void)n_in; (void)out_size; (void)ws_size;
}
